// DualLayerModel_26061861552146
// MI455X (gfx1250) — matrix operand order repaired and checked by reading, not compiled, not run on hardware
//
#include <hip/hip_runtime.h>
#include <hip/hip_bf16.h>
#include <math.h>

typedef __attribute__((ext_vector_type(16))) __bf16 v16bf;
typedef __attribute__((ext_vector_type(8)))  __bf16 v8bf;
typedef __attribute__((ext_vector_type(8)))  float  v8f;

#define GEMM_BM 128
#define GEMM_BN 128
#define GEMM_BK 64
#define LDS_STRIDE 72   // 64 + 8 pad (bf16): rows stay 16B aligned, banks conflict-free

// ---------------------------------------------------------------------------
// Tiled bf16 WMMA GEMM:  C[M,N] (f32) = A[M,K] (bf16, row-major) x Bt[N,K] (bf16)
// Block = 256 threads = 8 waves; wave grid 2(M) x 4(N); each wave -> 64x32 of C
// (4 M-subtiles x 2 N-subtiles = 8 WMMA tiles, 64 accum VGPRs/lane).
// Staging uses CDNA5 GLOBAL_LOAD_ASYNC_TO_LDS_B128 (ASYNCcnt) with LDS double
// buffering: the async copy of tile k+1 overlaps the 16 WMMAs of tile k.
// ---------------------------------------------------------------------------
__global__ __launch_bounds__(256)
void gemm_bf16_wmma_kernel(const __bf16* __restrict__ A,
                           const __bf16* __restrict__ Bt,
                           float* __restrict__ C,
                           int M, int Nn, int K)
{
    __shared__ __bf16 As[2][GEMM_BM][LDS_STRIDE];
    __shared__ __bf16 Bs[2][GEMM_BN][LDS_STRIDE];

    const int tid  = threadIdx.x;
    const int lane = tid & 31;
    const int wave = tid >> 5;
    const int wm   = wave & 1;   // 0..1 (M direction, 64 rows each)
    const int wn   = wave >> 1;  // 0..3 (N direction, 32 cols each)

    const int rowBase = blockIdx.y * GEMM_BM;
    const int colBase = blockIdx.x * GEMM_BN;

    // Staging: 128 rows x 64 bf16 each for A and B; 2 threads/row, 32 bf16 (64B) each.
    const int srow = tid >> 1;
    const int scol = (tid & 1) * 32;
    int agrow = rowBase + srow;
    if (agrow >= M) agrow = M - 1;            // clamp; C stores are guarded
    const int bgrow = colBase + srow;         // N is a multiple of 128: no guard

    v8f acc[4][2] = {};

    const int half = lane >> 4;               // 0 | 1
    const int l16  = lane & 15;

    // Async global->LDS copy of one 128x64 A tile + 128x64 B tile (4+4 b128/thread).
    // INST_OFFSET applies to both the global and LDS sides, so one address pair
    // covers all four 16B beats per tile half.
    auto issue_tile = [&](int buf, int k0) {
        unsigned la = (unsigned)(uintptr_t)&As[buf][srow][scol];
        unsigned lb = (unsigned)(uintptr_t)&Bs[buf][srow][scol];
        const __bf16* ga = A  + (size_t)agrow * K + k0 + scol;
        const __bf16* gb = Bt + (size_t)bgrow * K + k0 + scol;
        asm volatile(
            "global_load_async_to_lds_b128 %0, %2, off\n\t"
            "global_load_async_to_lds_b128 %0, %2, off offset:16\n\t"
            "global_load_async_to_lds_b128 %0, %2, off offset:32\n\t"
            "global_load_async_to_lds_b128 %0, %2, off offset:48\n\t"
            "global_load_async_to_lds_b128 %1, %3, off\n\t"
            "global_load_async_to_lds_b128 %1, %3, off offset:16\n\t"
            "global_load_async_to_lds_b128 %1, %3, off offset:32\n\t"
            "global_load_async_to_lds_b128 %1, %3, off offset:48"
            :: "v"(la), "v"(lb), "v"(ga), "v"(gb) : "memory");
    };

    issue_tile(0, 0);
    const int ktiles = K / GEMM_BK;

    for (int kt = 0; kt < ktiles; ++kt) {
        asm volatile("s_wait_asynccnt 0x0" ::: "memory");  // this wave's copies landed
        __syncthreads();                                   // all waves' tile kt visible
        if (kt + 1 < ktiles)
            issue_tile((kt + 1) & 1, (kt + 1) * GEMM_BK);  // overlaps with compute below
        const int buf = kt & 1;

#pragma unroll
        for (int ks = 0; ks < GEMM_BK; ks += 32) {
            // B fragments: lane holds column n = wn*32 + sn*16 + l16;
            // element e -> k = ks + 16*half + e  (32 contiguous bf16).
            v16bf bfrag[2];
#pragma unroll
            for (int sn = 0; sn < 2; ++sn) {
                const __bf16* bp = &Bs[buf][wn * 32 + sn * 16 + l16][ks + half * 16];
                v8bf lo = *(const v8bf*)bp;
                v8bf hi = *(const v8bf*)(bp + 8);
#pragma unroll
                for (int i = 0; i < 8; ++i) { bfrag[sn][i] = lo[i]; bfrag[sn][i + 8] = hi[i]; }
            }
            // A fragments: lane holds row m = wm*64 + sm*16 + l16;
            // element e -> k = ks + 16*half + e  (32 contiguous bf16).
            v16bf afrag[4];
#pragma unroll
            for (int sm = 0; sm < 4; ++sm) {
                const __bf16* ap = &As[buf][wm * 64 + sm * 16 + l16][ks + half * 16];
                v8bf lo = *(const v8bf*)ap;
                v8bf hi = *(const v8bf*)(ap + 8);
#pragma unroll
                for (int i = 0; i < 8; ++i) { afrag[sm][i] = lo[i]; afrag[sm][i + 8] = hi[i]; }
            }
#pragma unroll
            for (int sm = 0; sm < 4; ++sm)
#pragma unroll
                for (int sn = 0; sn < 2; ++sn)
                    acc[sm][sn] = __builtin_amdgcn_wmma_f32_16x16x32_bf16(
                        false, afrag[sm], false, bfrag[sn], (short)0,
                        acc[sm][sn], false, false);
        }
    }

    // C/D layout: VGPR r -> M = r + 8*half ; N = l16
#pragma unroll
    for (int sm = 0; sm < 4; ++sm) {
#pragma unroll
        for (int sn = 0; sn < 2; ++sn) {
            const int col = colBase + wn * 32 + sn * 16 + l16;
#pragma unroll
            for (int r = 0; r < 8; ++r) {
                int row = rowBase + wm * 64 + sm * 16 + half * 8 + r;
                if (row < M) C[(size_t)row * Nn + col] = acc[sm][sn][r];
            }
        }
    }
}

// ---------------------------------------------------------------------------
// Helpers / elementwise kernels
// ---------------------------------------------------------------------------
__device__ inline void block_reduce2_256(float& a, float& b)
{
    __shared__ float sA[8], sB[8];
#pragma unroll
    for (int m = 16; m >= 1; m >>= 1) {
        a += __shfl_xor(a, m, 32);
        b += __shfl_xor(b, m, 32);
    }
    const int wave = threadIdx.x >> 5;
    const int lane = threadIdx.x & 31;
    if (lane == 0) { sA[wave] = a; sB[wave] = b; }
    __syncthreads();
    a = 0.f; b = 0.f;
#pragma unroll
    for (int i = 0; i < 8; ++i) { a += sA[i]; b += sB[i]; }
    __syncthreads();
}

__global__ void f32_to_bf16_kernel(const float* __restrict__ in,
                                   __bf16* __restrict__ out, size_t n)
{
    size_t i = (size_t)blockIdx.x * blockDim.x + threadIdx.x;
    if (i < n) out[i] = (__bf16)in[i];
}

// Wt[n*K + k] = (bf16) W[k*N + n]
__global__ void transpose_to_bf16_kernel(const float* __restrict__ W,
                                         __bf16* __restrict__ Wt, int K, int N)
{
    size_t i = (size_t)blockIdx.x * blockDim.x + threadIdx.x;
    if (i >= (size_t)K * N) return;
    int k = (int)(i / N);
    int n = (int)(i % N);
    Wt[(size_t)n * K + k] = (__bf16)W[i];
}

__global__ void zero_f32_kernel(float* __restrict__ p, size_t n)
{
    size_t i = (size_t)blockIdx.x * blockDim.x + threadIdx.x;
    if (i < n) p[i] = 0.f;
}

// out[row, 0:D] = bf16(x[row]); out[row, D:2D] = bf16(agg[row])
__global__ void concat_bf16_kernel(const float* __restrict__ x,
                                   const float* __restrict__ agg,
                                   __bf16* __restrict__ out, int Nn, int D)
{
    size_t i = (size_t)blockIdx.x * blockDim.x + threadIdx.x;
    size_t total = (size_t)Nn * D;
    if (i >= total) return;
    size_t row = i / D, c = i % D;
    out[row * 2 * D + c]     = (__bf16)x[i];
    out[row * 2 * D + D + c] = (__bf16)agg[i];
}

// Per-edge: v = (1+softplus(coeff)*wt) * y[src] + pool_b ; LN ; relu ; scatter-max
// One wave per edge (D=256 -> 8 f32/lane). relu output >= 0 so integer-bit
// atomicMax == float max, and 0-init matches maximum(agg, 0).
__global__ __launch_bounds__(256)
void edge_pool_kernel(const float* __restrict__ y,
                      const int* __restrict__ me,       // [2,E]: src then dst
                      const float* __restrict__ ewt,
                      const float* __restrict__ pb,
                      const float* __restrict__ g,
                      const float* __restrict__ b,
                      const float* __restrict__ coeffp,
                      int* __restrict__ aggBits, int E)
{
    const int D = 256;
    const int wave = threadIdx.x >> 5;
    const int lane = threadIdx.x & 31;
    const int e = blockIdx.x * 8 + wave;
    if (e >= E) return;
    const int src = me[e];
    const int dst = me[E + e];
    const float coeff = log1pf(expf(coeffp[0]));       // softplus
    const float sc = 1.0f + coeff * ewt[e];

    const float* yrow = y + (size_t)src * D;
    float v[8]; float s = 0.f, s2 = 0.f;
#pragma unroll
    for (int j = 0; j < 8; ++j) {
        int c = j * 32 + lane;
        float x = sc * yrow[c] + pb[c];
        v[j] = x; s += x; s2 += x * x;
    }
#pragma unroll
    for (int m = 16; m >= 1; m >>= 1) {
        s  += __shfl_xor(s, m, 32);
        s2 += __shfl_xor(s2, m, 32);
    }
    const float mean = s / D;
    const float var  = s2 / D - mean * mean;
    const float rstd = rsqrtf(var + 1e-5f);
    int* arow = aggBits + (size_t)dst * D;
#pragma unroll
    for (int j = 0; j < 8; ++j) {
        int c = j * 32 + lane;
        float xn = fmaxf((v[j] - mean) * rstd * g[c] + b[c], 0.f);
        atomicMax(&arow[c], __float_as_int(xn));
    }
}

// Row kernel: v = pre + bias ; LN(g,b) ; optional relu ; optional f32/bf16 out.
__global__ __launch_bounds__(256)
void bias_ln_act_kernel(const float* __restrict__ pre,
                        const float* __restrict__ bias,
                        const float* __restrict__ g,
                        const float* __restrict__ b,
                        float* __restrict__ outF,
                        __bf16* __restrict__ outB,
                        int W, int do_relu)
{
    const int row = blockIdx.x;
    const int tid = threadIdx.x;
    const int per = W >> 8;                 // W/256 in {1,2,4,8}
    const float* prow = pre + (size_t)row * W;
    float v[8]; float s = 0.f, s2 = 0.f;
    for (int j = 0; j < per; ++j) {
        int c = j * 256 + tid;
        float x = prow[c] + bias[c];
        v[j] = x; s += x; s2 += x * x;
    }
    block_reduce2_256(s, s2);
    const float inv  = 1.0f / (float)W;
    const float mean = s * inv;
    const float var  = s2 * inv - mean * mean;
    const float rstd = rsqrtf(var + 1e-5f);
    for (int j = 0; j < per; ++j) {
        int c = j * 256 + tid;
        float xn = (v[j] - mean) * rstd * g[c] + b[c];
        if (do_relu) xn = fmaxf(xn, 0.f);
        if (outF) outF[(size_t)row * W + c] = xn;
        if (outB) outB[(size_t)row * W + c] = (__bf16)xn;
    }
}

// Decoder front: e = LN0(concat(a+b, a*b)) -> bf16 [S, 512]
__global__ __launch_bounds__(256)
void edge_combine_ln_kernel(const float* __restrict__ h,
                            const int* __restrict__ sup,   // [2,S]
                            const float* __restrict__ g,
                            const float* __restrict__ b,
                            __bf16* __restrict__ outB, int S)
{
    const int D = 256, W = 512;
    const int r = blockIdx.x;
    const int tid = threadIdx.x;
    const int ea = sup[r];
    const int eb = sup[S + r];
    const float av = h[(size_t)ea * D + tid];
    const float bv = h[(size_t)eb * D + tid];
    const float v0 = av + bv;
    const float v1 = av * bv;
    float s = v0 + v1, s2 = v0 * v0 + v1 * v1;
    block_reduce2_256(s, s2);
    const float mean = s / W;
    const float var  = s2 / W - mean * mean;
    const float rstd = rsqrtf(var + 1e-5f);
    outB[(size_t)r * W + tid]     = (__bf16)((v0 - mean) * rstd * g[tid] + b[tid]);
    outB[(size_t)r * W + D + tid] = (__bf16)((v1 - mean) * rstd * g[tid + D] + b[tid + D]);
}

// Fused final LN+relu + dual linear heads (N=1 each): probs / relu(wts)
__global__ __launch_bounds__(256)
void heads_kernel(const float* __restrict__ pre,
                  const float* __restrict__ bias,
                  const float* __restrict__ g,
                  const float* __restrict__ b,
                  const float* __restrict__ pW, const float* __restrict__ pB,
                  const float* __restrict__ wW, const float* __restrict__ wB,
                  float* __restrict__ out, int S)
{
    const int W = 1024;
    const int r = blockIdx.x;
    const int tid = threadIdx.x;
    const float* prow = pre + (size_t)r * W;
    float v[4]; float s = 0.f, s2 = 0.f;
#pragma unroll
    for (int j = 0; j < 4; ++j) {
        int c = j * 256 + tid;
        float x = prow[c] + bias[c];
        v[j] = x; s += x; s2 += x * x;
    }
    block_reduce2_256(s, s2);
    const float mean = s / W;
    const float var  = s2 / W - mean * mean;
    const float rstd = rsqrtf(var + 1e-5f);
    float dp = 0.f, dw = 0.f;
#pragma unroll
    for (int j = 0; j < 4; ++j) {
        int c = j * 256 + tid;
        float e = fmaxf((v[j] - mean) * rstd * g[c] + b[c], 0.f);
        dp += e * pW[c];
        dw += e * wW[c];
    }
    block_reduce2_256(dp, dw);
    if (tid == 0) {
        out[r]     = dp + pB[0];
        out[S + r] = fmaxf(dw + wB[0], 0.f);
    }
}

// ---------------------------------------------------------------------------
// Host orchestration
// ---------------------------------------------------------------------------
static inline void launch_gemm(const __bf16* A, const __bf16* Bt, float* C,
                               int M, int Nn, int K, hipStream_t stream)
{
    dim3 grid(Nn / GEMM_BN, (M + GEMM_BM - 1) / GEMM_BM);
    gemm_bf16_wmma_kernel<<<grid, 256, 0, stream>>>(A, Bt, C, M, Nn, K);
}

extern "C" void kernel_launch(void* const* d_in, const int* in_sizes, int n_in,
                              void* d_out, int out_size, void* d_ws, size_t ws_size,
                              hipStream_t stream)
{
    (void)n_in; (void)out_size; (void)ws_size;
    const int D  = 256;
    const int Nn = in_sizes[0] / D;   // 50000 nodes
    const int S  = in_sizes[1] / 2;   // 100000 supervision edges
    const int E  = in_sizes[2] / 2;   // 800000 message edges

    // -------- inputs (jax pytree: dict keys sorted; tuples/lists in order) --
    const float* x    = (const float*)d_in[0];
    const int*   sup  = (const int*)d_in[1];
    const int*   me   = (const int*)d_in[2];
    const float* ewt  = (const float*)d_in[3];
    // conv1: coeff, fin_W, fin_b, lnf_b, lnf_g, lnp_b, lnp_g, pool_W, pool_b
    const float* c1_coeff = (const float*)d_in[4];
    const float* c1_finW  = (const float*)d_in[5];
    const float* c1_finb  = (const float*)d_in[6];
    const float* c1_lnfb  = (const float*)d_in[7];
    const float* c1_lnfg  = (const float*)d_in[8];
    const float* c1_lnpb  = (const float*)d_in[9];
    const float* c1_lnpg  = (const float*)d_in[10];
    const float* c1_poolW = (const float*)d_in[11];
    const float* c1_poolb = (const float*)d_in[12];
    const float* c2_coeff = (const float*)d_in[13];
    const float* c2_finW  = (const float*)d_in[14];
    const float* c2_finb  = (const float*)d_in[15];
    const float* c2_lnfb  = (const float*)d_in[16];
    const float* c2_lnfg  = (const float*)d_in[17];
    const float* c2_lnpb  = (const float*)d_in[18];
    const float* c2_lnpg  = (const float*)d_in[19];
    const float* c2_poolW = (const float*)d_in[20];
    const float* c2_poolb = (const float*)d_in[21];
    // dec: ln0_b, ln0_g, mlp[(W,b,g,be)x4], prob_W, prob_b, wt_W, wt_b
    const float* ln0b = (const float*)d_in[22];
    const float* ln0g = (const float*)d_in[23];
    const float* mW1 = (const float*)d_in[24]; const float* mb1 = (const float*)d_in[25];
    const float* mg1 = (const float*)d_in[26]; const float* me1 = (const float*)d_in[27];
    const float* mW2 = (const float*)d_in[28]; const float* mb2 = (const float*)d_in[29];
    const float* mg2 = (const float*)d_in[30]; const float* me2 = (const float*)d_in[31];
    const float* mW3 = (const float*)d_in[32]; const float* mb3 = (const float*)d_in[33];
    const float* mg3 = (const float*)d_in[34]; const float* me3 = (const float*)d_in[35];
    const float* mW4 = (const float*)d_in[36]; const float* mb4 = (const float*)d_in[37];
    const float* mg4 = (const float*)d_in[38]; const float* me4 = (const float*)d_in[39];
    const float* pW  = (const float*)d_in[40]; const float* pB  = (const float*)d_in[41];
    const float* wW  = (const float*)d_in[42]; const float* wB  = (const float*)d_in[43];
    float* out = (float*)d_out;

    // -------- workspace carve-up ------------------------------------------
    size_t off = 0;
    auto alloc = [&](size_t bytes) -> char* {
        off = (off + 255) & ~(size_t)255;
        char* p = (char*)d_ws + off;
        off += bytes;
        return p;
    };
    __bf16* pWt1 = (__bf16*)alloc((size_t)256 * 256 * 2);
    __bf16* fWt1 = (__bf16*)alloc((size_t)256 * 512 * 2);
    __bf16* pWt2 = (__bf16*)alloc((size_t)256 * 256 * 2);
    __bf16* fWt2 = (__bf16*)alloc((size_t)256 * 512 * 2);
    __bf16* W1t  = (__bf16*)alloc((size_t)2048 * 512 * 2);
    __bf16* W2t  = (__bf16*)alloc((size_t)2048 * 2048 * 2);
    __bf16* W3t  = (__bf16*)alloc((size_t)1024 * 2048 * 2);
    __bf16* W4t  = (__bf16*)alloc((size_t)1024 * 1024 * 2);
    __bf16* abuf = (__bf16*)alloc((size_t)S * 2048 * 2);   // ping (also conv A/cat)
    __bf16* bbuf = (__bf16*)alloc((size_t)S * 2048 * 2);   // pong
    float*  pre  = (float*) alloc((size_t)S * 2048 * 4);   // all GEMM outputs
    float*  agg  = (float*) alloc((size_t)Nn * D * 4);
    float*  h1   = (float*) alloc((size_t)Nn * D * 4);
    float*  h2   = (float*) alloc((size_t)Nn * D * 4);

    auto egrid = [](size_t n) { return (unsigned)((n + 255) / 256); };

    // -------- weight prep: fp32 [K,N] -> bf16 [N,K] ------------------------
    transpose_to_bf16_kernel<<<egrid(256 * 256), 256, 0, stream>>>(c1_poolW, pWt1, 256, 256);
    transpose_to_bf16_kernel<<<egrid(512 * 256), 256, 0, stream>>>(c1_finW,  fWt1, 512, 256);
    transpose_to_bf16_kernel<<<egrid(256 * 256), 256, 0, stream>>>(c2_poolW, pWt2, 256, 256);
    transpose_to_bf16_kernel<<<egrid(512 * 256), 256, 0, stream>>>(c2_finW,  fWt2, 512, 256);
    transpose_to_bf16_kernel<<<egrid((size_t)512 * 2048),  256, 0, stream>>>(mW1, W1t, 512, 2048);
    transpose_to_bf16_kernel<<<egrid((size_t)2048 * 2048), 256, 0, stream>>>(mW2, W2t, 2048, 2048);
    transpose_to_bf16_kernel<<<egrid((size_t)2048 * 1024), 256, 0, stream>>>(mW3, W3t, 2048, 1024);
    transpose_to_bf16_kernel<<<egrid((size_t)1024 * 1024), 256, 0, stream>>>(mW4, W4t, 1024, 1024);

    const size_t nd = (size_t)Nn * D;

    // -------- conv1: x -> h1 ----------------------------------------------
    f32_to_bf16_kernel<<<egrid(nd), 256, 0, stream>>>(x, abuf, nd);
    launch_gemm(abuf, pWt1, pre, Nn, 256, 256, stream);            // y = x @ poolW
    zero_f32_kernel<<<egrid(nd), 256, 0, stream>>>(agg, nd);
    edge_pool_kernel<<<(E + 7) / 8, 256, 0, stream>>>(pre, me, ewt, c1_poolb,
                                                      c1_lnpg, c1_lnpb, c1_coeff,
                                                      (int*)agg, E);
    concat_bf16_kernel<<<egrid(nd), 256, 0, stream>>>(x, agg, abuf, Nn, D);
    launch_gemm(abuf, fWt1, pre, Nn, 256, 512, stream);            // concat @ finW
    bias_ln_act_kernel<<<Nn, 256, 0, stream>>>(pre, c1_finb, c1_lnfg, c1_lnfb,
                                               h1, abuf, 256, 1);  // h1 f32 + bf16

    // -------- conv2: h1 -> h2 ---------------------------------------------
    launch_gemm(abuf, pWt2, pre, Nn, 256, 256, stream);
    zero_f32_kernel<<<egrid(nd), 256, 0, stream>>>(agg, nd);
    edge_pool_kernel<<<(E + 7) / 8, 256, 0, stream>>>(pre, me, ewt, c2_poolb,
                                                      c2_lnpg, c2_lnpb, c2_coeff,
                                                      (int*)agg, E);
    concat_bf16_kernel<<<egrid(nd), 256, 0, stream>>>(h1, agg, abuf, Nn, D);
    launch_gemm(abuf, fWt2, pre, Nn, 256, 512, stream);
    bias_ln_act_kernel<<<Nn, 256, 0, stream>>>(pre, c2_finb, c2_lnfg, c2_lnfb,
                                               h2, nullptr, 256, 1);

    // -------- decoder ------------------------------------------------------
    edge_combine_ln_kernel<<<S, 256, 0, stream>>>(h2, sup, ln0g, ln0b, abuf, S);
    launch_gemm(abuf, W1t, pre, S, 2048, 512, stream);
    bias_ln_act_kernel<<<S, 256, 0, stream>>>(pre, mb1, mg1, me1, nullptr, bbuf, 2048, 1);
    launch_gemm(bbuf, W2t, pre, S, 2048, 2048, stream);
    bias_ln_act_kernel<<<S, 256, 0, stream>>>(pre, mb2, mg2, me2, nullptr, abuf, 2048, 1);
    launch_gemm(abuf, W3t, pre, S, 1024, 2048, stream);
    bias_ln_act_kernel<<<S, 256, 0, stream>>>(pre, mb3, mg3, me3, nullptr, bbuf, 1024, 1);
    launch_gemm(bbuf, W4t, pre, S, 1024, 1024, stream);
    heads_kernel<<<S, 256, 0, stream>>>(pre, mb4, mg4, me4, pW, pB, wW, wB, out, S);
}
